// HINGE5_56985626083400
// MI455X (gfx1250) — matrix operand order repaired and checked by reading, not compiled, not run on hardware
//
#include <hip/hip_runtime.h>
#include <hip/hip_bf16.h>

typedef __attribute__((ext_vector_type(16))) _Float16 v16h;
typedef __attribute__((ext_vector_type(8)))  float    v8f;

#define NF    200
#define NFP   208             // NF padded to 16
#define EMB   128
#define JOUT  126             // EMB - 2
#define BATCH 2048
#define MTOT  (BATCH * JOUT)  // 258048
#define MTILES (MTOT / 16)    // 16128
#define WPB   8               // waves per block in passB
#define NTILES 13             // 208/16
#define CNT   258048.0f
#define EPS_F 1e-5f

// ---------------- fragment loaders (CDNA5 16-bit WMMA layouts) ----------------
// A 16x32 f16: lanes 0-15 -> M=lane, K base 0; lanes 16-31 -> M=lane-16, K base 8.
// per-lane element e -> K = base + e + (e>=8 ? 8 : 0)
__device__ __forceinline__ v16h load_a_frag(const _Float16* tile /*[16][32]*/) {
    int lane = threadIdx.x & 31;
    int m  = lane & 15;
    int bk = (lane >> 4) * 8;
    v16h a;
#pragma unroll
    for (int e = 0; e < 16; ++e) {
        int k = bk + e + (e >= 8 ? 8 : 0);
        a[e] = tile[m * 32 + k];
    }
    return a;
}

// B 32xN panel slice: lanes 0-15 K base 0, lanes 16-31 K base 8; element e -> K = base + e + (e>=8 ? 8 : 0); N = nf0 + lane%16.
__device__ __forceinline__ v16h load_b_frag(const _Float16* panel /*[32][NFP]*/, int nf0) {
    int lane = threadIdx.x & 31;
    int n  = nf0 + (lane & 15);
    int kb = (lane >> 4) * 8;
    v16h b;
#pragma unroll
    for (int e = 0; e < 16; ++e) {
        b[e] = panel[(kb + e + (e >= 8 ? 8 : 0)) * NFP + n];
    }
    return b;
}

// ---------------- kernel 0: init workspace + output ----------------
__global__ void k_init(float* stats, float* out, const float* fc_b) {
    int i = blockIdx.x * blockDim.x + threadIdx.x;
    if (i < 405)   stats[i] = 0.0f;
    if (i < BATCH) out[i]   = fc_b[0];
}

// ---------------- kernel 1: tap sums and second moments ----------------
// stats layout per k (k=0..2), stride 135:
//   [0..14]  : sum over (b,j) of x2k[t]
//   [15..134]: sum of x2k[t1]*x2k[t2] for t1<=t2 (packed upper triangle, 120)
__global__ __launch_bounds__(512)
void k_stats(const int* __restrict__ x_batch,
             const float* __restrict__ emb_roles,
             const float* __restrict__ emb_values,
             float* __restrict__ stats) {
    __shared__ float rows[9][EMB];
    int b = blockIdx.x;
    const int* xb = x_batch + b * 10;
    int t = threadIdx.x;

    for (int idx = t; idx < 9 * EMB; idx += 512) {
        int rr = idx >> 7, c = idx & 127;
        const float* p;
        switch (rr) {
            case 0: p = emb_values + (size_t)xb[1] * EMB; break;  // hrt0 = fv0
            case 1: p = emb_roles  + (size_t)xb[0] * EMB; break;  // hrt1 = fr0
            case 2: p = emb_values + (size_t)xb[3] * EMB; break;  // hrt2 = fv1
            case 3: p = emb_roles  + (size_t)xb[4] * EMB; break;  // kr0
            case 4: p = emb_values + (size_t)xb[5] * EMB; break;  // kv0
            case 5: p = emb_roles  + (size_t)xb[6] * EMB; break;  // kr1
            case 6: p = emb_values + (size_t)xb[7] * EMB; break;  // kv1
            case 7: p = emb_roles  + (size_t)xb[8] * EMB; break;  // kr2
            default: p = emb_values + (size_t)xb[9] * EMB; break; // kv2
        }
        rows[rr][c] = p[c];
    }
    __syncthreads();

    if (t < 405) {
        int k = t / 135, le = t % 135;
        float acc = 0.0f;
        if (le < 15) {
            int t1 = le, rs = t1 / 3, o1 = t1 % 3;
            int rI = (rs < 3) ? rs : ((rs == 3) ? 3 + 2 * k : 4 + 2 * k);
            for (int j = 0; j < JOUT; ++j) acc += rows[rI][j + o1];
        } else {
            int pe = le - 15;
            int t1 = 0;
            while (pe >= 15 - t1) { pe -= 15 - t1; ++t1; }
            int t2 = t1 + pe;
            int r1s = t1 / 3, o1 = t1 % 3;
            int r2s = t2 / 3, o2 = t2 % 3;
            int rI1 = (r1s < 3) ? r1s : ((r1s == 3) ? 3 + 2 * k : 4 + 2 * k);
            int rI2 = (r2s < 3) ? r2s : ((r2s == 3) ? 3 + 2 * k : 4 + 2 * k);
            for (int j = 0; j < JOUT; ++j) acc += rows[rI1][j + o1] * rows[rI2][j + o2];
        }
        atomicAdd(&stats[t], acc);
    }
}

// ---------------- kernel 2: stats -> per-filter affine (scale, shift) ----------------
__global__ __launch_bounds__(640)
void k_finalize(const float* __restrict__ stats,
                const float* __restrict__ c1w, const float* __restrict__ c1b,
                const float* __restrict__ g1,  const float* __restrict__ be1,
                const float* __restrict__ c2w, const float* __restrict__ c2b,
                const float* __restrict__ g2,  const float* __restrict__ be2,
                float* scale1, float* shift1, float* scale2, float* shift2) {
    int t = threadIdx.x;
    auto S = [&](int k, int a, int b2) -> float {
        if (a > b2) { int tmp = a; a = b2; b2 = tmp; }
        int p = a * 15 - (a * (a - 1)) / 2 + (b2 - a);
        return stats[k * 135 + 15 + p];
    };
    if (t < NF) {   // conv1 (k=0 top-left 9x9 block; hrt taps identical across k)
        int f = t;
        float w[9];
#pragma unroll
        for (int i = 0; i < 9; ++i) w[i] = c1w[f * 9 + i];
        float sw = 0.0f;
#pragma unroll
        for (int i = 0; i < 9; ++i) sw += w[i] * stats[i];
        float bias = c1b[f];
        float mean = (sw + CNT * bias) / CNT;
        float q = 0.0f;
        for (int i = 0; i < 9; ++i)
            for (int jj = 0; jj < 9; ++jj) q += w[i] * w[jj] * S(0, i, jj);
        float ey2 = (q + 2.0f * bias * sw + CNT * bias * bias) / CNT;
        float var = ey2 - mean * mean;
        float sc  = g1[f] * rsqrtf(var + EPS_F);
        scale1[f] = sc;
        shift1[f] = be1[f] + (bias - mean) * sc;
    }
    if (t < 3 * NF) {   // conv2, per (k, f)
        int k = t / NF, f = t % NF;
        float w[15];
#pragma unroll
        for (int i = 0; i < 15; ++i) w[i] = c2w[f * 15 + i];
        float sw = 0.0f;
#pragma unroll
        for (int i = 0; i < 15; ++i) sw += w[i] * stats[k * 135 + i];
        float bias = c2b[f];
        float mean = (sw + CNT * bias) / CNT;
        float q = 0.0f;
        for (int i = 0; i < 15; ++i)
            for (int jj = 0; jj < 15; ++jj) q += w[i] * w[jj] * S(k, i, jj);
        float ey2 = (q + 2.0f * bias * sw + CNT * bias * bias) / CNT;
        float var = ey2 - mean * mean;
        float sc  = g2[f] * rsqrtf(var + EPS_F);
        scale2[k * NF + f] = sc;
        shift2[k * NF + f] = be2[f] + (bias - mean) * sc;
    }
}

// ---------------- kernel 3: fused WMMA conv1+conv2 + BN + ReLU + min + fc ----------------
// 8 waves/block; each wave owns one 16-row M-tile. B panels staged once per block.
__global__ __launch_bounds__(256)
void k_passB(const int* __restrict__ x_batch,
             const float* __restrict__ emb_roles,
             const float* __restrict__ emb_values,
             const float* __restrict__ c1w,
             const float* __restrict__ c2w,
             const float* __restrict__ scale1, const float* __restrict__ shift1,
             const float* __restrict__ scale2, const float* __restrict__ shift2,
             const float* __restrict__ fcw,
             float* __restrict__ out) {
    __shared__ _Float16 Bw1[32 * NFP];          // conv1 weights^T, K zero-padded
    __shared__ _Float16 Bw2[32 * NFP];          // conv2 weights^T
    __shared__ _Float16 Atl[WPB][4][16 * 32];   // per-wave A tiles: conv1 + 3x conv2

    int t    = threadIdx.x;
    int w    = t >> 5;
    int lane = t & 31;

    // ---- stage both B panels once (whole block) ----
    for (int idx = t; idx < 32 * NFP; idx += 256) {
        int kk = idx / NFP, f = idx % NFP;
        float w1 = 0.0f, w2 = 0.0f;
        if (f < NF) {
            if (kk < 9)  w1 = c1w[f * 9  + kk];
            if (kk < 15) w2 = c2w[f * 15 + kk];
        }
        Bw1[idx] = (_Float16)w1;
        Bw2[idx] = (_Float16)w2;
    }

    // ---- build this wave's f16 A tiles (im2col, K zero-padded to 32) ----
    int row0 = (blockIdx.x * WPB + w) * 16;
    {
        int m  = lane & 15;
        int kh = lane >> 4;                       // K-half this lane fills
        int r = row0 + m;
        int b = r / JOUT, j = r % JOUT;
        const int* xb = x_batch + b * 10;
        const float* r0 = emb_values + (size_t)xb[1] * EMB;
        const float* r1 = emb_roles  + (size_t)xb[0] * EMB;
        const float* r2 = emb_values + (size_t)xb[3] * EMB;
#pragma unroll
        for (int e = 0; e < 16; ++e) {
            int kk = kh * 16 + e;
            float v = 0.0f;
            if (kk < 9) {
                const float* p = (kk < 3) ? r0 : (kk < 6) ? r1 : r2;
                v = p[j + kk % 3];
            }
            Atl[w][0][m * 32 + kk] = (_Float16)v;
        }
        for (int k = 0; k < 3; ++k) {
            const float* r3 = emb_roles  + (size_t)xb[4 + 2 * k] * EMB;
            const float* r4 = emb_values + (size_t)xb[5 + 2 * k] * EMB;
#pragma unroll
            for (int e = 0; e < 16; ++e) {
                int kk = kh * 16 + e;
                float v = 0.0f;
                if (kk < 15) {
                    int rs = kk / 3;
                    const float* p = (rs == 0) ? r0 : (rs == 1) ? r1 : (rs == 2) ? r2
                                    : (rs == 3) ? r3 : r4;
                    v = p[j + kk % 3];
                }
                Atl[w][1 + k][m * 32 + kk] = (_Float16)v;
            }
        }
    }
    __syncthreads();   // one barrier: B panels + all A tiles resident

    v16h a1  = load_a_frag(Atl[w][0]);
    v16h a20 = load_a_frag(Atl[w][1]);
    v16h a21 = load_a_frag(Atl[w][2]);
    v16h a22 = load_a_frag(Atl[w][3]);

    int n  = lane & 15;
    int mh = (lane >> 4) * 8;

    int jr[8], br[8];
#pragma unroll
    for (int i = 0; i < 8; ++i) {
        int r = row0 + mh + i;
        br[i] = r / JOUT;
        jr[i] = r % JOUT;
    }

    float acc[8] = {0.f, 0.f, 0.f, 0.f, 0.f, 0.f, 0.f, 0.f};

    for (int nt = 0; nt < NTILES; ++nt) {
        int nf0 = nt * 16;
        v16h b1 = load_b_frag(Bw1, nf0);
        v16h b2 = load_b_frag(Bw2, nf0);
        v8f z = {0.f, 0.f, 0.f, 0.f, 0.f, 0.f, 0.f, 0.f};
        v8f c1  = __builtin_amdgcn_wmma_f32_16x16x32_f16(false, a1,  false, b1, (short)0, z, false, false);
        v8f c20 = __builtin_amdgcn_wmma_f32_16x16x32_f16(false, a20, false, b2, (short)0, z, false, false);
        v8f c21 = __builtin_amdgcn_wmma_f32_16x16x32_f16(false, a21, false, b2, (short)0, z, false, false);
        v8f c22 = __builtin_amdgcn_wmma_f32_16x16x32_f16(false, a22, false, b2, (short)0, z, false, false);

        int f = nf0 + n;
        if (f < NF) {
            float s1  = scale1[f],          h1  = shift1[f];
            float s20 = scale2[0 * NF + f], h20 = shift2[0 * NF + f];
            float s21 = scale2[1 * NF + f], h21 = shift2[1 * NF + f];
            float s22 = scale2[2 * NF + f], h22 = shift2[2 * NF + f];
#pragma unroll
            for (int i = 0; i < 8; ++i) {
                float y1 = fmaxf(fmaf(c1[i],  s1,  h1),  0.0f);
                float q0 = fmaxf(fmaf(c20[i], s20, h20), 0.0f);
                float q1 = fmaxf(fmaf(c21[i], s21, h21), 0.0f);
                float q2 = fmaxf(fmaf(c22[i], s22, h22), 0.0f);
                float mn = fminf(fminf(y1, q0), fminf(q1, q2));
                acc[i] += mn * fcw[f * JOUT + jr[i]];
            }
        }
    }

    // ---- reduce across the 16 N-lanes with lane permutes (no LDS) ----
#pragma unroll
    for (int mask = 1; mask < 16; mask <<= 1) {
#pragma unroll
        for (int i = 0; i < 8; ++i) acc[i] += __shfl_xor(acc[i], mask, 16);
    }
    if (n == 0) {
#pragma unroll
        for (int i = 0; i < 8; ++i) atomicAdd(&out[br[i]], acc[i]);
    }
}

// ---------------- host launcher ----------------
extern "C" void kernel_launch(void* const* d_in, const int* in_sizes, int n_in,
                              void* d_out, int out_size, void* d_ws, size_t ws_size,
                              hipStream_t stream) {
    (void)in_sizes; (void)n_in; (void)out_size; (void)ws_size;
    const int*   x_batch   = (const int*)d_in[0];
    const float* emb_roles = (const float*)d_in[3];
    const float* emb_vals  = (const float*)d_in[4];
    const float* c1w = (const float*)d_in[5];
    const float* c1b = (const float*)d_in[6];
    const float* g1  = (const float*)d_in[7];
    const float* be1 = (const float*)d_in[8];
    const float* c2w = (const float*)d_in[9];
    const float* c2b = (const float*)d_in[10];
    const float* g2  = (const float*)d_in[11];
    const float* be2 = (const float*)d_in[12];
    const float* fcw = (const float*)d_in[13];
    const float* fcb = (const float*)d_in[14];
    float* out = (float*)d_out;

    float* ws     = (float*)d_ws;
    float* stats  = ws;          // 405 floats
    float* scale1 = ws + 512;    // 200
    float* shift1 = ws + 768;    // 200
    float* scale2 = ws + 1024;   // 600
    float* shift2 = ws + 1664;   // 600

    k_init<<<8, 256, 0, stream>>>(stats, out, fcb);
    k_stats<<<BATCH, 512, 0, stream>>>(x_batch, emb_roles, emb_vals, stats);
    k_finalize<<<1, 640, 0, stream>>>(stats, c1w, c1b, g1, be1,
                                      c2w, c2b, g2, be2,
                                      scale1, shift1, scale2, shift2);
    k_passB<<<MTILES / WPB, 32 * WPB, 0, stream>>>(x_batch, emb_roles, emb_vals, c1w, c2w,
                                                   scale1, shift1, scale2, shift2, fcw, out);
}
